// MultipoleNetRes_2010044694543
// MI455X (gfx1250) — matrix operand order repaired and checked by reading, not compiled, not run on hardware
//
#include <hip/hip_runtime.h>

// MultipoleNet on gfx1250: all dense layers via V_WMMA_F32_16X16X32_F16,
// fp16 activation storage for edge features, fp32 accumulation + fp32 atomics
// for segment sums and residuals.

#define NNODES 10000
#define NEDGES 320000

typedef _Float16 h16v __attribute__((ext_vector_type(16)));
typedef _Float16 h8v  __attribute__((ext_vector_type(8)));
typedef __fp16   pk2v __attribute__((ext_vector_type(2)));  // cvt_pkrtz result
typedef float    f8v  __attribute__((ext_vector_type(8)));
typedef float    f2v  __attribute__((ext_vector_type(2)));

// mila(x) = x * tanh(softplus(x-1)). With t = exp(x-1):
//   tanh(log(1+t)) = ((1+t)^2 - 1)/((1+t)^2 + 1) = (t^2 + 2t)/(t^2 + 2t + 2)
// Branch-free, no tanhf/log1pf: 1 transcendental + a few VALU ops.
__device__ __forceinline__ float mila_f(float x) {
  float y = fminf(x - 1.0f, 20.0f);   // clamp avoids inf/inf; ratio==1 there
  float t = __expf(y);
  float u = t * (t + 2.0f);
  return x * u * __builtin_amdgcn_rcpf(u + 2.0f);
}

// Packed f32->f16 row conversion (v_cvt_pk_f16_f32 + 32-bit stores).
template <int N>
__device__ __forceinline__ void cvt_row(_Float16* dst, const float* src) {
#pragma unroll
  for (int k = 0; k < N; k += 2) {
    f2v s = *(const f2v*)(src + k);
    *(pk2v*)(dst + k) = __builtin_amdgcn_cvt_pkrtz(s.x, s.y);
  }
}

// ---- WMMA fragment helpers (layouts per CDNA5 ISA 7.12.2, wave32) ----------
// A (16x32 f16): lane m=l&15; halves 0..7 -> K = kb + (l>=16?8:0) + 0..7
//                            halves 8..15 -> K = kb + 16 + (l>=16?8:0) + 0..7
__device__ __forceinline__ h16v afrag_lds(const _Float16* base, int ldk, int kb) {
  const int lane = threadIdx.x & 31;
  const int m = lane & 15;
  const int khi = (lane >> 4) << 3;
  const _Float16* p = base + m * ldk + kb + khi;
  h8v lo = *(const h8v*)(p);
  h8v hi = *(const h8v*)(p + 16);
  return __builtin_shufflevector(lo, hi, 0,1,2,3,4,5,6,7,8,9,10,11,12,13,14,15);
}

// B (32x16 f16): lane l: N = l&15; half h -> K = (l>=16?8:0) + (h<8 ? h : h+8).
// Weights pre-swizzled so each fragment is 16 contiguous halves per lane.
__device__ __forceinline__ h16v bfrag_lds(const _Float16* w, int f) {
  const int lane = threadIdx.x & 31;
  return *(const h16v*)(w + ((f * 32 + lane) << 4));
}

__device__ __forceinline__ f8v wmma_f16(h16v a, h16v b, f8v c) {
  return __builtin_amdgcn_wmma_f32_16x16x32_f16(false, a, false, b, (short)0, c,
                                                false, false);
}

// D = A[16 x 32*KT] * W + bias, N=64 (4 n-tiles). A staged in LDS (f16).
template <int KT>
__device__ __forceinline__ void gemm_tile(const _Float16* aLds, int ldk,
                                          const _Float16* wLds,
                                          const float* bias, f8v acc[4]) {
  const int lane = threadIdx.x & 31;
  const int n0 = lane & 15;
#pragma unroll
  for (int nt = 0; nt < 4; ++nt) {
    float bv = bias[nt * 16 + n0];
#pragma unroll
    for (int i = 0; i < 8; ++i) acc[nt][i] = bv;
  }
#pragma unroll
  for (int kt = 0; kt < KT; ++kt) {
    h16v a = afrag_lds(aLds, ldk, kt * 32);
#pragma unroll
    for (int nt = 0; nt < 4; ++nt)
      acc[nt] = wmma_f16(a, bfrag_lds(wLds, kt * 4 + nt), acc[nt]);
  }
}

// C layout: lane l: N = nt*16 + (l&15); reg r -> M = r + (l>=16?8:0)
__device__ __forceinline__ void store_tile_mila_lds(const f8v acc[4],
                                                    _Float16* out, int ldk) {
  const int lane = threadIdx.x & 31;
  const int n0 = lane & 15;
  const int mh = (lane >> 4) << 3;
#pragma unroll
  for (int nt = 0; nt < 4; ++nt) {
#pragma unroll
    for (int r = 0; r < 8; ++r)
      out[(mh + r) * ldk + nt * 16 + n0] = (_Float16)mila_f(acc[nt][r]);
  }
}

__device__ __forceinline__ void copy_w(_Float16* dst, const _Float16* src,
                                       int halves) {
  for (int i = threadIdx.x * 8; i < halves; i += 128 * 8)
    *(h8v*)(dst + i) = *(const h8v*)(src + i);
}

// 64-wide dot of an f16 LDS row against f32 weights (vector LDS reads).
__device__ __forceinline__ float dot64_lds(const _Float16* h, const float* w) {
  float s = 0.0f;
#pragma unroll
  for (int k = 0; k < 64; k += 8) {
    h8v v = *(const h8v*)(h + k);
#pragma unroll
    for (int j = 0; j < 8; ++j) s += (float)v[j] * w[k + j];
  }
  return s;
}

// ---- repack fp32 row-major [din][dout] weights into B-fragment layout ------
__global__ void k_repack(const float* __restrict__ w, _Float16* __restrict__ out,
                         int din, int dout, int KT, int NT) {
  int idx = blockIdx.x * blockDim.x + threadIdx.x;
  int total = KT * NT * 512;
  if (idx >= total) return;
  int h = idx & 15;
  int lane = (idx >> 4) & 31;
  int f = idx >> 9;
  int kt = f / NT, nt = f % NT;
  int K = kt * 32 + ((lane >= 16) ? 8 : 0) + (h < 8 ? h : h + 8);
  int Nc = nt * 16 + (lane & 15);
  float v = (K < din && Nc < dout) ? w[(size_t)K * dout + Nc] : 0.0f;
  out[idx] = (_Float16)v;
}

__global__ void k_zero(float* p, int n) {
  int i = blockIdx.x * blockDim.x + threadIdx.x;
  if (i < n) p[i] = 0.0f;
}

// ---- embeddings -------------------------------------------------------------
__global__ __launch_bounds__(128) void k_embed_node(
    const float* __restrict__ nodes, const _Float16* __restrict__ wf,
    const float* __restrict__ bias, float* __restrict__ nf, int N) {
  __shared__ _Float16 sW[2048];
  __shared__ _Float16 sA[4][512];
  copy_w(sW, wf, 2048);
  int wave = threadIdx.x >> 5, lane = threadIdx.x & 31;
  int tile = blockIdx.x * 4 + wave;
  int m = lane & 15;
  int row = tile * 16 + m;
  int rc = row < N ? row : 0;
  _Float16* arow = &sA[wave][m * 32];
  if (lane < 16) {
#pragma unroll
    for (int k = 0; k < 16; ++k)
      arow[k] = (k < 7) ? (_Float16)nodes[(size_t)rc * 7 + k] : (_Float16)0.0f;
  } else {
#pragma unroll
    for (int k = 16; k < 32; ++k) arow[k] = (_Float16)0.0f;
  }
  __syncthreads();
  f8v acc[4];
  gemm_tile<1>(&sA[wave][0], 32, sW, bias, acc);
  int n0 = lane & 15, mh = (lane >> 4) << 3;
#pragma unroll
  for (int nt = 0; nt < 4; ++nt) {
#pragma unroll
    for (int r = 0; r < 8; ++r) {
      int rr = tile * 16 + mh + r;
      if (rr < N) nf[(size_t)rr * 64 + nt * 16 + n0] = mila_f(acc[nt][r]);
    }
  }
}

__global__ __launch_bounds__(128) void k_embed_edge(
    const float* __restrict__ edges, const _Float16* __restrict__ wf,
    const float* __restrict__ bias, _Float16* __restrict__ ef, int E) {
  __shared__ _Float16 sW[2048];
  __shared__ _Float16 sA[4][512];
  copy_w(sW, wf, 2048);
  int wave = threadIdx.x >> 5, lane = threadIdx.x & 31;
  int tile = blockIdx.x * 4 + wave;
  int m = lane & 15;
  long e = (long)tile * 16 + m;
  long ec = e < E ? e : 0;
  _Float16* arow = &sA[wave][m * 32];
  if (lane < 16) {
    cvt_row<16>(arow, edges + ec * 32);
  } else {
    cvt_row<16>(arow + 16, edges + ec * 32 + 16);
  }
  __syncthreads();
  f8v acc[4];
  gemm_tile<1>(&sA[wave][0], 32, sW, bias, acc);
  int n0 = lane & 15, mh = (lane >> 4) << 3;
#pragma unroll
  for (int nt = 0; nt < 4; ++nt) {
#pragma unroll
    for (int r = 0; r < 8; ++r) {
      long rr = (long)tile * 16 + mh + r;
      if (rr < E) ef[rr * 64 + nt * 16 + n0] = (_Float16)mila_f(acc[nt][r]);
    }
  }
}

// ---- edge block: ef' = mila2(concat(ef, nf[r], nf[s])); agg[r] += ef' ------
__global__ __launch_bounds__(128) void k_edge_block(
    const _Float16* __restrict__ ef_in, _Float16* __restrict__ ef_out,
    const float* __restrict__ nf, float* __restrict__ agg,
    const int* __restrict__ senders, const int* __restrict__ receivers,
    const _Float16* __restrict__ w1f, const float* __restrict__ b1,
    const _Float16* __restrict__ w2f, const float* __restrict__ b2, int E) {
  __shared__ _Float16 sW1[12288];  // 192->64
  __shared__ _Float16 sW2[4096];   // 64->64
  __shared__ _Float16 sA[4][3072]; // 16 rows x 192 halves per wave
  copy_w(sW1, w1f, 12288);
  copy_w(sW2, w2f, 4096);
  int wave = threadIdx.x >> 5, lane = threadIdx.x & 31;
  int tile = blockIdx.x * 4 + wave;
  long e0 = (long)tile * 16;
  int m = lane & 15;
  long e = e0 + m;
  int ri = receivers[e], si = senders[e];
  _Float16* row = &sA[wave][m * 192];
  if (lane < 16) {
    const h8v* src = (const h8v*)(ef_in + e * 64);
    h8v* dst = (h8v*)row;
#pragma unroll
    for (int i = 0; i < 8; ++i) dst[i] = src[i];
    cvt_row<32>(row + 64, nf + (size_t)ri * 64);
  } else {
    cvt_row<32>(row + 96, nf + (size_t)ri * 64 + 32);
    cvt_row<64>(row + 128, nf + (size_t)si * 64);
  }
  __syncthreads();
  f8v acc[4];
  gemm_tile<6>(&sA[wave][0], 192, sW1, b1, acc);
  _Float16* sH = &sA[wave][0];  // reuse wave-private LDS as 16x64 h1 tile
  store_tile_mila_lds(acc, sH, 64);
  __builtin_amdgcn_wave_barrier();
  f8v acc2[4];
  gemm_tile<2>(sH, 64, sW2, b2, acc2);
  int n0 = lane & 15, mh = (lane >> 4) << 3;
#pragma unroll
  for (int r = 0; r < 8; ++r) {
    long er = e0 + mh + r;
    int rr = receivers[er];
#pragma unroll
    for (int nt = 0; nt < 4; ++nt) {
      float v = mila_f(acc2[nt][r]);
      ef_out[er * 64 + nt * 16 + n0] = (_Float16)v;
      atomicAdd(&agg[(size_t)rr * 64 + nt * 16 + n0], v);
    }
  }
}

// ---- node block: nf += mila2(concat(agg, nf)) ------------------------------
__global__ __launch_bounds__(128) void k_node_block(
    const float* __restrict__ agg, float* __restrict__ nf,
    const _Float16* __restrict__ w1f, const float* __restrict__ b1,
    const _Float16* __restrict__ w2f, const float* __restrict__ b2, int N) {
  __shared__ _Float16 sW1[8192];   // 128->64
  __shared__ _Float16 sW2[4096];   // 64->64
  __shared__ _Float16 sA[4][2048]; // 16 x 128
  copy_w(sW1, w1f, 8192);
  copy_w(sW2, w2f, 4096);
  int wave = threadIdx.x >> 5, lane = threadIdx.x & 31;
  int tile = blockIdx.x * 4 + wave;
  int m = lane & 15;
  int row = tile * 16 + m;
  int rc = row < N ? row : 0;
  _Float16* arow = &sA[wave][m * 128];
  if (lane < 16) {
    cvt_row<64>(arow, agg + (size_t)rc * 64);
  } else {
    cvt_row<64>(arow + 64, nf + (size_t)rc * 64);
  }
  __syncthreads();
  f8v acc[4];
  gemm_tile<4>(&sA[wave][0], 128, sW1, b1, acc);
  _Float16* sH = &sA[wave][0];
  store_tile_mila_lds(acc, sH, 64);
  __builtin_amdgcn_wave_barrier();
  f8v acc2[4];
  gemm_tile<2>(sH, 64, sW2, b2, acc2);
  int n0 = lane & 15, mh = (lane >> 4) << 3;
#pragma unroll
  for (int nt = 0; nt < 4; ++nt) {
#pragma unroll
    for (int r = 0; r < 8; ++r) {
      int rr = tile * 16 + mh + r;
      if (rr < N) {
        size_t idx = (size_t)rr * 64 + nt * 16 + n0;
        nf[idx] += mila_f(acc2[nt][r]);
      }
    }
  }
}

// ---- heads -----------------------------------------------------------------
__global__ __launch_bounds__(128) void k_mono_head(
    const float* __restrict__ nmono, const _Float16* __restrict__ w0f,
    const float* __restrict__ b0, const _Float16* __restrict__ w1f,
    const float* __restrict__ b1, const float* __restrict__ w2,
    const float* __restrict__ b2, float* __restrict__ tmp,
    float* __restrict__ sum, int N) {
  __shared__ _Float16 sW0[4096];
  __shared__ _Float16 sW1[4096];
  __shared__ _Float16 sA[4][1024];
  copy_w(sW0, w0f, 4096);
  copy_w(sW1, w1f, 4096);
  int wave = threadIdx.x >> 5, lane = threadIdx.x & 31;
  int tile = blockIdx.x * 4 + wave;
  int m = lane & 15;
  int row = tile * 16 + m;
  int rc = row < N ? row : 0;
  _Float16* arow = &sA[wave][m * 64];
  const float* s = nmono + (size_t)rc * 64;
  if (lane < 16) {
    cvt_row<32>(arow, s);
  } else {
    cvt_row<32>(arow + 32, s + 32);
  }
  __syncthreads();
  f8v acc[4];
  gemm_tile<2>(&sA[wave][0], 64, sW0, b0, acc);
  _Float16* sH = &sA[wave][0];
  store_tile_mila_lds(acc, sH, 64);
  __builtin_amdgcn_wave_barrier();
  f8v acc2[4];
  gemm_tile<2>(sH, 64, sW1, b1, acc2);
  store_tile_mila_lds(acc2, sH, 64);
  __builtin_amdgcn_wave_barrier();
  if (lane < 16 && row < N) {
    float sv = b2[0] + dot64_lds(sH + lane * 64, w2);
    tmp[row] = sv;
    atomicAdd(sum, sv);
  }
}

__global__ void k_mono_final(const float* __restrict__ tmp,
                             const float* __restrict__ sum,
                             float* __restrict__ out, int N) {
  int i = blockIdx.x * blockDim.x + threadIdx.x;
  if (i < N) out[i] = tmp[i] - (*sum) / (float)N;
}

// mode 1: dipoles [N,3]; mode 2: quadrupoles [N,3,3]
__global__ __launch_bounds__(128) void k_edge_head(
    const float* __restrict__ ntw, const float* __restrict__ edges,
    const float* __restrict__ coords, const int* __restrict__ senders,
    const int* __restrict__ receivers, const _Float16* __restrict__ w0f,
    const float* __restrict__ b0, const _Float16* __restrict__ w1f,
    const float* __restrict__ b1, const float* __restrict__ w2,
    const float* __restrict__ b2, float* __restrict__ out, int E, int mode) {
  __shared__ _Float16 sW0[10240]; // 160->64
  __shared__ _Float16 sW1[4096];  // 64->64
  __shared__ _Float16 sA[4][2560];
  copy_w(sW0, w0f, 10240);
  copy_w(sW1, w1f, 4096);
  int wave = threadIdx.x >> 5, lane = threadIdx.x & 31;
  int tile = blockIdx.x * 4 + wave;
  int m = lane & 15;
  long e = (long)tile * 16 + m;
  int si = senders[e], ri = receivers[e];
  _Float16* row = &sA[wave][m * 160];
  if (lane < 16) {
    cvt_row<64>(row, ntw + (size_t)si * 64);
    cvt_row<16>(row + 64, ntw + (size_t)ri * 64);
  } else {
    cvt_row<48>(row + 80, ntw + (size_t)ri * 64 + 16);
    cvt_row<32>(row + 128, edges + e * 32);
  }
  __syncthreads();
  f8v acc[4];
  gemm_tile<5>(&sA[wave][0], 160, sW0, b0, acc);
  _Float16* sH = &sA[wave][0];
  store_tile_mila_lds(acc, sH, 64);
  __builtin_amdgcn_wave_barrier();
  f8v acc2[4];
  gemm_tile<2>(sH, 64, sW1, b1, acc2);
  store_tile_mila_lds(acc2, sH, 64);
  __builtin_amdgcn_wave_barrier();
  if (lane < 16) {
    float w = b2[0] + dot64_lds(sH + lane * 64, w2);
    float vx = coords[(size_t)si * 3 + 0] - coords[(size_t)ri * 3 + 0];
    float vy = coords[(size_t)si * 3 + 1] - coords[(size_t)ri * 3 + 1];
    float vz = coords[(size_t)si * 3 + 2] - coords[(size_t)ri * 3 + 2];
    if (mode == 1) {
      atomicAdd(&out[(size_t)ri * 3 + 0], vx * w);
      atomicAdd(&out[(size_t)ri * 3 + 1], vy * w);
      atomicAdd(&out[(size_t)ri * 3 + 2], vz * w);
    } else {
      float tr3 = (vx * vx + vy * vy + vz * vz) / 3.0f;
      float o[9] = {vx * vx - tr3, vx * vy, vx * vz,
                    vy * vx, vy * vy - tr3, vy * vz,
                    vz * vx, vz * vy, vz * vz - tr3};
#pragma unroll
      for (int j = 0; j < 9; ++j) atomicAdd(&out[(size_t)ri * 9 + j], o[j] * w);
    }
  }
}

// ---- host orchestration -----------------------------------------------------
struct DenseOff { size_t b, w; int din, dout; };
struct TowerOff {
  DenseOff emb_edge, emb_node, e1[4], e2[4], n1[4], n2[4], out0, out1, out2;
};

// jax tree-flatten order: dict keys sorted; {"b","w"} -> b then w.
static void walk_tower(size_t& cur, int out_in, TowerOff& T) {
  auto dense = [&](int din, int dout) {
    DenseOff d; d.b = cur; cur += (size_t)dout; d.w = cur;
    cur += (size_t)din * dout; d.din = din; d.dout = dout; return d;
  };
  T.emb_edge = dense(32, 64);
  T.emb_node = dense(7, 64);
  for (int i = 0; i < 4; ++i) {
    T.e1[i] = dense(192, 64);
    T.e2[i] = dense(64, 64);
    T.n1[i] = dense(128, 64);
    T.n2[i] = dense(64, 64);
  }
  T.out0 = dense(out_in, 64);
  T.out1 = dense(64, 64);
  T.out2 = dense(64, 1);
}

struct Frags { _Float16 *emb_e, *emb_n, *e1[4], *e2[4], *n1[4], *n2[4], *o0, *o1; };

extern "C" void kernel_launch(void* const* d_in, const int* in_sizes, int n_in,
                              void* d_out, int out_size, void* d_ws,
                              size_t ws_size, hipStream_t stream) {
  (void)in_sizes; (void)n_in; (void)out_size; (void)ws_size;
  const float* nodes = (const float*)d_in[0];
  const float* edges = (const float*)d_in[1];
  const float* coords = (const float*)d_in[2];
  const float* P = (const float*)d_in[3];  // flat params (sorted-key tree order)
  const int* senders = (const int*)d_in[4];
  const int* receivers = (const int*)d_in[5];
  float* out = (float*)d_out;

  TowerOff Td, Tm, Tq;
  size_t cur = 0;
  walk_tower(cur, 160, Td);  // "dipo"
  walk_tower(cur, 64, Tm);   // "mono"
  walk_tower(cur, 160, Tq);  // "quad"

  char* ws = (char*)d_ws;
  size_t wcur = 0;
  auto walloc = [&](size_t bytes) -> char* {
    size_t o = (wcur + 255) & ~(size_t)255;
    wcur = o + bytes;
    return ws + o;
  };

  auto repack = [&](const DenseOff& d) -> _Float16* {
    int KT = (d.din + 31) / 32, NT = (d.dout + 15) / 16;
    int total = KT * NT * 512;
    _Float16* p = (_Float16*)walloc((size_t)total * 2);
    k_repack<<<(total + 255) / 256, 256, 0, stream>>>(P + d.w, p, d.din, d.dout,
                                                      KT, NT);
    return p;
  };
  auto repack_tower = [&](const TowerOff& T) -> Frags {
    Frags F;
    F.emb_e = repack(T.emb_edge);
    F.emb_n = repack(T.emb_node);
    for (int i = 0; i < 4; ++i) {
      F.e1[i] = repack(T.e1[i]);
      F.e2[i] = repack(T.e2[i]);
      F.n1[i] = repack(T.n1[i]);
      F.n2[i] = repack(T.n2[i]);
    }
    F.o0 = repack(T.out0);
    F.o1 = repack(T.out1);
    return F;
  };
  Frags Fd = repack_tower(Td), Fm = repack_tower(Tm), Fq = repack_tower(Tq);

  float* nf_m = (float*)walloc((size_t)NNODES * 64 * 4);
  float* nf_d = (float*)walloc((size_t)NNODES * 64 * 4);
  float* nf_q = (float*)walloc((size_t)NNODES * 64 * 4);
  _Float16* efA = (_Float16*)walloc((size_t)NEDGES * 64 * 2);
  _Float16* efB = (_Float16*)walloc((size_t)NEDGES * 64 * 2);
  float* agg = (float*)walloc((size_t)NNODES * 64 * 4);
  float* mtmp = (float*)walloc((size_t)NNODES * 4);
  float* msum = (float*)walloc(256);

  const int NODE_TILES = (NNODES + 15) / 16;    // 625
  const int NODE_BLOCKS = (NODE_TILES + 3) / 4; // 157
  const int EDGE_BLOCKS = (NEDGES / 16) / 4;    // 5000

  auto run_tower = [&](const TowerOff& T, const Frags& F, float* nf_t) {
    k_embed_node<<<NODE_BLOCKS, 128, 0, stream>>>(nodes, F.emb_n,
                                                  P + T.emb_node.b, nf_t, NNODES);
    k_embed_edge<<<EDGE_BLOCKS, 128, 0, stream>>>(edges, F.emb_e,
                                                  P + T.emb_edge.b, efA, NEDGES);
    _Float16* ein = efA;
    _Float16* eout = efB;
    for (int s = 0; s < 4; ++s) {
      k_zero<<<(NNODES * 64 + 255) / 256, 256, 0, stream>>>(agg, NNODES * 64);
      k_edge_block<<<EDGE_BLOCKS, 128, 0, stream>>>(
          ein, eout, nf_t, agg, senders, receivers, F.e1[s], P + T.e1[s].b,
          F.e2[s], P + T.e2[s].b, NEDGES);
      k_node_block<<<NODE_BLOCKS, 128, 0, stream>>>(
          agg, nf_t, F.n1[s], P + T.n1[s].b, F.n2[s], P + T.n2[s].b, NNODES);
      _Float16* t = ein; ein = eout; eout = t;
    }
  };
  run_tower(Tm, Fm, nf_m);
  run_tower(Td, Fd, nf_d);
  run_tower(Tq, Fq, nf_q);

  // outputs: mono[10000] | dipoles[10000*3] | quadrupoles[10000*9]
  k_zero<<<(130000 + 255) / 256, 256, 0, stream>>>(out, 130000);
  k_zero<<<1, 32, 0, stream>>>(msum, 1);
  k_mono_head<<<NODE_BLOCKS, 128, 0, stream>>>(
      nf_m, Fm.o0, P + Tm.out0.b, Fm.o1, P + Tm.out1.b, P + Tm.out2.w,
      P + Tm.out2.b, mtmp, msum, NNODES);
  k_mono_final<<<(NNODES + 255) / 256, 256, 0, stream>>>(mtmp, msum, out, NNODES);
  k_edge_head<<<EDGE_BLOCKS, 128, 0, stream>>>(
      nf_d, edges, coords, senders, receivers, Fd.o0, P + Td.out0.b, Fd.o1,
      P + Td.out1.b, P + Td.out2.w, P + Td.out2.b, out + NNODES, NEDGES, 1);
  k_edge_head<<<EDGE_BLOCKS, 128, 0, stream>>>(
      nf_q, edges, coords, senders, receivers, Fq.o0, P + Tq.out0.b, Fq.o1,
      P + Tq.out1.b, P + Tq.out2.w, P + Tq.out2.b, out + NNODES + 3 * NNODES,
      NEDGES, 2);
}
